// NeuroSATAssign_23373212025281
// MI455X (gfx1250) — matrix operand order repaired and checked by reading, not compiled, not run on hardware
//
#include <hip/hip_runtime.h>
#include <hip/hip_bf16.h>

// ---------------- problem constants ----------------
#define BB   4
#define LLIT 2048
#define NCL  4096
#define DD   128
#define TT   1024      // LLIT/2
#define CC2  256       // 2*DD
#define G4   512       // 4*DD
#define ITERS 4

typedef __attribute__((ext_vector_type(16))) __bf16 v16bf;
typedef __attribute__((ext_vector_type(8)))  float  v8f;

union bfu {
    v16bf v;
    unsigned int u[8];
    __bf16 h[16];
};

union q4 {
    unsigned long long q;
    __bf16 h[4];
};

#define WMMA_BF16(A_, B_, C_) \
    __builtin_amdgcn_wmma_f32_16x16x32_bf16(false, (A_), false, (B_), (short)0, (C_), false, false)

// ---------------------------------------------------------------------------
// bf16 WMMA GEMM:  C = act(A @ W + bias)   (all matrix data bf16, bias f32)
//   A: [M x K] bf16 (row major; transA: element (m,k) at A[k*lda + m])
//   W: [K x N] bf16 row major
//   C: [M x N] bf16 row major
// Workgroup tile 128x128, 8 waves (4x2), each wave owns 32x64 (8 accumulators).
// Requires M%128==0, N%128==0, K%32==0 (true for every call below).
// ---------------------------------------------------------------------------
__global__ __launch_bounds__(256)
void gemm_wmma(const __bf16* __restrict__ A, long long aB, int lda, int transA,
               const __bf16* __restrict__ W, long long wB, int ldw,
               const float* __restrict__ bias,
               __bf16* __restrict__ C, long long cB, int ldc,
               int M, int N, int K, int act)
{
    __shared__ __bf16 As[128][36];   // [m][k], pitch 36 halves keeps b64 stores aligned
    __shared__ __bf16 Bs[128][36];   // [n][k] (W tile stored transposed)

    const int tid  = threadIdx.x;
    const int wv   = tid >> 5;
    const int lane = tid & 31;
    const int ln   = lane & 15;
    const int g    = lane >> 4;
    const int wm   = wv & 3;        // 4 row-groups of 32
    const int wn   = wv >> 2;       // 2 col-groups of 64
    const int m0   = blockIdx.y * 128;
    const int n0   = blockIdx.x * 128;

    const __bf16* Ab = A + aB * (long long)blockIdx.z;
    const __bf16* Wb = W + wB * (long long)blockIdx.z;
    __bf16*       Cb = C + cB * (long long)blockIdx.z;

    v8f zeroc = {};
    v8f acc[2][4];
    #pragma unroll
    for (int i = 0; i < 2; ++i)
        #pragma unroll
        for (int j = 0; j < 4; ++j) acc[i][j] = zeroc;

    for (int k0 = 0; k0 < K; k0 += 32) {
        // ---- stage A tile (128x32 halves) ----
        if (transA) {
            for (int c = tid; c < 1024; c += 256) {
                int k = c >> 5, mc = (c & 31) * 4;               // contiguous along m
                const q4* gp = (const q4*)(Ab + (long long)(k0 + k) * lda + (m0 + mc));
                q4 v = *gp;
                if (k0 + 32 < K) __builtin_prefetch((const char*)gp + (long long)32 * lda * 2, 0, 1);
                #pragma unroll
                for (int j2 = 0; j2 < 4; ++j2) As[mc + j2][k] = v.h[j2];
            }
        } else {
            for (int c = tid; c < 1024; c += 256) {
                int m = c >> 3, kc = (c & 7) * 4;                // contiguous along k
                const q4* gp = (const q4*)(Ab + (long long)(m0 + m) * lda + (k0 + kc));
                q4 v = *gp;
                if (k0 + 32 < K) __builtin_prefetch((const char*)gp + 64, 0, 1);
                *(q4*)&As[m][kc] = v;
            }
        }
        // ---- stage W tile (32x128) transposed -> Bs[n][k] ----
        for (int c = tid; c < 1024; c += 256) {
            int k = c >> 5, nc = (c & 31) * 4;
            const q4* gp = (const q4*)(Wb + (long long)(k0 + k) * ldw + (n0 + nc));
            q4 v = *gp;
            if (k0 + 32 < K) __builtin_prefetch((const char*)gp + (long long)32 * ldw * 2, 0, 1);
            #pragma unroll
            for (int j2 = 0; j2 < 4; ++j2) Bs[nc + j2][k] = v.h[j2];
        }
        __syncthreads();

        // ---- fragments ----
        bfu af[2], bfr[4];
        #pragma unroll
        for (int i = 0; i < 2; ++i) {
            const unsigned* arow = (const unsigned*)&As[wm * 32 + i * 16 + ln][0];
            #pragma unroll
            for (int v = 0; v < 8; ++v) {
                int k2 = g * 16 + v * 2;   // A layout: k = g*16+h
                af[i].u[v] = arow[k2 >> 1];
            }
        }
        #pragma unroll
        for (int j = 0; j < 4; ++j) {
            const unsigned* brow = (const unsigned*)&Bs[wn * 64 + j * 16 + ln][0];
            #pragma unroll
            for (int v = 0; v < 8; ++v) bfr[j].u[v] = brow[g * 8 + v];   // B layout: k = g*16+h
        }
        #pragma unroll
        for (int i = 0; i < 2; ++i)
            #pragma unroll
            for (int j = 0; j < 4; ++j)
                acc[i][j] = WMMA_BF16(af[i].v, bfr[j].v, acc[i][j]);
        __syncthreads();
    }

    // ---- epilogue ----
    #pragma unroll
    for (int j = 0; j < 4; ++j) {
        int gn = n0 + wn * 64 + j * 16 + ln;
        float bv = bias ? bias[gn] : 0.0f;
        #pragma unroll
        for (int i = 0; i < 2; ++i) {
            #pragma unroll
            for (int r = 0; r < 8; ++r) {
                int gm = m0 + wm * 32 + i * 16 + r + 8 * g;
                float v = acc[i][j][r] + bv;
                if (act == 1) v = fmaxf(v, 0.0f);
                Cb[(long long)gm * ldc + gn] = (__bf16)v;
            }
        }
    }
}

// ---------------------------------------------------------------------------
// Fused LayerNorm-LSTM pointwise. One wave per row. bf16 gates/h, f32 cell.
// ---------------------------------------------------------------------------
__global__ __launch_bounds__(256)
void lstm_ln(const __bf16* __restrict__ G1, const __bf16* __restrict__ G2,
             const float* __restrict__ gig, const float* __restrict__ gib,
             const float* __restrict__ ghg, const float* __restrict__ ghb,
             const float* __restrict__ gcg, const float* __restrict__ gcb,
             const float* __restrict__ Cprev,
             __bf16* __restrict__ Hout, float* __restrict__ Cout)
{
    const int wv = threadIdx.x >> 5;
    const int lane = threadIdx.x & 31;
    const long long row = (long long)blockIdx.x * 8 + wv;
    const __bf16* g1 = G1 + row * G4;
    const __bf16* g2 = G2 + row * G4;

    float a1[16], a2[16];
    float s1 = 0.f, s2 = 0.f, q1 = 0.f, q2 = 0.f;
    #pragma unroll
    for (int j = 0; j < 16; ++j) {
        a1[j] = (float)g1[lane + 32 * j];
        a2[j] = (float)g2[lane + 32 * j];
        s1 += a1[j]; q1 += a1[j] * a1[j];
        s2 += a2[j]; q2 += a2[j] * a2[j];
    }
    #pragma unroll
    for (int m = 1; m < 32; m <<= 1) {
        s1 += __shfl_xor(s1, m, 32); q1 += __shfl_xor(q1, m, 32);
        s2 += __shfl_xor(s2, m, 32); q2 += __shfl_xor(q2, m, 32);
    }
    const float inv512 = 1.0f / 512.0f;
    float m1 = s1 * inv512, m2 = s2 * inv512;
    float v1 = q1 * inv512 - m1 * m1, v2 = q2 * inv512 - m2 * m2;
    float r1 = rsqrtf(v1 + 1e-5f), r2 = rsqrtf(v2 + 1e-5f);

    float gate[16];
    #pragma unroll
    for (int j = 0; j < 16; ++j) {
        int p = lane + 32 * j;
        gate[j] = (a1[j] - m1) * r1 * gig[p] + gib[p]
                + (a2[j] - m2) * r2 * ghg[p] + ghb[p];
    }

    float cpre[4];
    float sc = 0.f, qc = 0.f;
    #pragma unroll
    for (int j = 0; j < 4; ++j) {
        int d = lane + 32 * j;
        float ii = 1.0f / (1.0f + __expf(-gate[j]));
        float ff = 1.0f / (1.0f + __expf(-gate[j + 4]));
        float gg = tanhf(gate[j + 8]);
        cpre[j] = ff * Cprev[row * DD + d] + ii * gg;
        sc += cpre[j]; qc += cpre[j] * cpre[j];
    }
    #pragma unroll
    for (int m = 1; m < 32; m <<= 1) {
        sc += __shfl_xor(sc, m, 32); qc += __shfl_xor(qc, m, 32);
    }
    float mc = sc / 128.0f;
    float vc = qc / 128.0f - mc * mc;
    float rc = rsqrtf(vc + 1e-5f);
    #pragma unroll
    for (int j = 0; j < 4; ++j) {
        int d = lane + 32 * j;
        float cn = (cpre[j] - mc) * rc * gcg[d] + gcb[d];
        float oo = 1.0f / (1.0f + __expf(-gate[j + 12]));
        Cout[row * DD + d] = cn;
        Hout[row * DD + d] = (__bf16)(oo * tanhf(cn));
    }
}

// ---------------------------------------------------------------------------
// Flash attention on bf16 Q/K/V, one wave per 16-row q tile.
// ---------------------------------------------------------------------------
#define ATTN_WAVES 4
__global__ __launch_bounds__(128)
void attn_flash(const __bf16* __restrict__ Q, const __bf16* __restrict__ Kd,
                const __bf16* __restrict__ V, const int* __restrict__ cnt,
                __bf16* __restrict__ Y)
{
    __shared__ __bf16 pslab[ATTN_WAVES][256];   // per-wave 16x16 P bounce

    const int wv = threadIdx.x >> 5;
    const int lane = threadIdx.x & 31;
    const int ln = lane & 15;
    const int g  = lane >> 4;
    const int bb = blockIdx.z;
    const int hh = blockIdx.y;
    const int t0 = (blockIdx.x * ATTN_WAVES + wv) * 16;
    const int hf = cnt[bb] >> 1;
    const float scale = 0.17677669529663687f;   // 1/sqrt(32)

    // Q A-fragment: bf16 pairs are contiguous -> direct u32 loads
    bfu qf;
    {
        const __bf16* qrow = Q + ((long long)(bb * TT + t0 + ln)) * CC2 + hh * 32;
        #pragma unroll
        for (int v = 0; v < 8; ++v) {
            int k = g * 16 + 2 * v;
            qf.u[v] = *(const unsigned*)(qrow + k);
        }
    }

    float mrun[8], lrun[8];
    #pragma unroll
    for (int r = 0; r < 8; ++r) { mrun[r] = -1e30f; lrun[r] = 0.0f; }
    v8f o0 = {};
    v8f o1 = {};

    for (int s0 = 0; s0 < TT; s0 += 16) {
        // K^T B-fragment: column n = key s0+ln, K-index = head dim g*16+h
        bfu kf;
        {
            const __bf16* krow = Kd + ((long long)(bb * TT + s0 + ln)) * CC2 + hh * 32;
            #pragma unroll
            for (int v = 0; v < 8; ++v)
                kf.u[v] = *(const unsigned*)(krow + g * 16 + 2 * v);
        }
        v8f zeroc = {};
        v8f s = WMMA_BF16(qf.v, kf.v, zeroc);

        const bool colvalid = (s0 + ln) < hf;
        float p[8];
        #pragma unroll
        for (int r = 0; r < 8; ++r) {
            float sv = s[r] * scale;
            if (!colvalid) sv = -1e30f;
            float rmax = sv;
            #pragma unroll
            for (int m = 1; m < 16; m <<= 1) rmax = fmaxf(rmax, __shfl_xor(rmax, m, 32));
            float nm   = fmaxf(mrun[r], rmax);
            float corr = __expf(mrun[r] - nm);
            mrun[r] = nm;
            float pv = colvalid ? __expf(sv - nm) : 0.0f;
            float rsum = pv;
            #pragma unroll
            for (int m = 1; m < 16; m <<= 1) rsum += __shfl_xor(rsum, m, 32);
            lrun[r] = lrun[r] * corr + rsum;
            o0[r] *= corr;
            o1[r] *= corr;
            p[r] = pv;
        }

        // transpose P (C layout -> A layout) through LDS
        #pragma unroll
        for (int r = 0; r < 8; ++r)
            pslab[wv][(r + 8 * g) * 16 + ln] = (__bf16)p[r];
        asm volatile("s_wait_dscnt 0" ::: "memory");

        bfu pf;
        {
            const unsigned* pu = (const unsigned*)&pslab[wv][0];
            #pragma unroll
            for (int v = 0; v < 8; ++v) pf.u[v] = (g == 0) ? pu[ln * 8 + v] : 0u;  // k = g*16+2v(+1); g=1 half is K pad 16 -> 32
        }

        // V B-fragments (K-index = s within tile; k>=16 half-wave zero pad)
        bfu vf0, vf1;
        #pragma unroll
        for (int v = 0; v < 8; ++v) {
            long long r0 = ((long long)(bb * TT + s0 + 2 * v)) * CC2 + hh * 32;
            long long r1 = ((long long)(bb * TT + s0 + 2 * v + 1)) * CC2 + hh * 32;
            vf0.h[2 * v]     = (g == 0) ? V[r0 + ln]      : (__bf16)0.0f;
            vf0.h[2 * v + 1] = (g == 0) ? V[r1 + ln]      : (__bf16)0.0f;
            vf1.h[2 * v]     = (g == 0) ? V[r0 + 16 + ln] : (__bf16)0.0f;
            vf1.h[2 * v + 1] = (g == 0) ? V[r1 + 16 + ln] : (__bf16)0.0f;
        }
        o0 = WMMA_BF16(pf.v, vf0.v, o0);
        o1 = WMMA_BF16(pf.v, vf1.v, o1);
    }

    #pragma unroll
    for (int r = 0; r < 8; ++r) {
        int trow = t0 + r + 8 * g;
        float invl = (lrun[r] > 0.0f) ? (1.0f / lrun[r]) : 0.0f;
        long long base = ((long long)(bb * TT + trow)) * CC2 + hh * 32;
        Y[base + ln]      = (__bf16)(o0[r] * invl);
        Y[base + 16 + ln] = (__bf16)(o1[r] * invl);
    }
}

// ---------------------------------------------------------------------------
// Small pointwise kernels
// ---------------------------------------------------------------------------
__global__ __launch_bounds__(256)
void cvt_f32_bf16(const float* __restrict__ s, __bf16* __restrict__ d, long long n)
{
    long long i = (long long)blockIdx.x * 256 + threadIdx.x;
    if (i < n) d[i] = (__bf16)s[i];
}

__global__ __launch_bounds__(256)
void init_state(const float* __restrict__ Li, const float* __restrict__ Ci,
                __bf16* __restrict__ Lh, float* __restrict__ Lc,
                __bf16* __restrict__ Ch, float* __restrict__ Cc)
{
    int i = blockIdx.x * 256 + threadIdx.x;
    if (i < BB * LLIT * DD) { Lh[i] = (__bf16)Li[i % DD]; Lc[i] = 0.0f; }
    if (i < BB * NCL * DD)  { Ch[i] = (__bf16)Ci[i % DD]; Cc[i] = 0.0f; }
}

// X = [CL ; flip(Lh)] per literal row
__global__ __launch_bounds__(256)
void build_xl(const __bf16* __restrict__ CLmsg, const __bf16* __restrict__ Lh,
              __bf16* __restrict__ X)
{
    int i = blockIdx.x * 256 + threadIdx.x;           // over BB*LLIT*DD
    int d = i % DD;
    int row = i / DD;
    int l = row % LLIT;
    int b = row / LLIT;
    int lf = l ^ (LLIT / 2);                           // swap pos/neg halves
    X[(long long)row * CC2 + d]      = CLmsg[(long long)row * DD + d];
    X[(long long)row * CC2 + DD + d] = Lh[((long long)(b * LLIT + lf)) * DD + d];
}

// read = mask * [Lh[:, :T] ; Lh[:, T:2T]]
__global__ __launch_bounds__(256)
void build_read(const __bf16* __restrict__ Lh, const int* __restrict__ cnt,
                __bf16* __restrict__ R)
{
    int i = blockIdx.x * 256 + threadIdx.x;           // over BB*TT*DD
    int d = i % DD;
    int row = i / DD;
    int t = row % TT;
    int b = row / TT;
    bool m = t < (cnt[b] >> 1);
    R[(long long)row * CC2 + d]      = m ? Lh[((long long)(b * LLIT + t)) * DD + d]      : (__bf16)0.0f;
    R[(long long)row * CC2 + DD + d] = m ? Lh[((long long)(b * LLIT + t + TT)) * DD + d] : (__bf16)0.0f;
}

// out[row] = mask * sigmoid(dot(H[row], W2) + b2), one wave per row
__global__ __launch_bounds__(256)
void head_out(const __bf16* __restrict__ H, const float* __restrict__ W2,
              const float* __restrict__ b2, const int* __restrict__ cnt,
              float* __restrict__ out)
{
    int wv = threadIdx.x >> 5;
    int lane = threadIdx.x & 31;
    int row = blockIdx.x * 8 + wv;
    const __bf16* hrow = H + (long long)row * DD;
    float s = 0.0f;
    #pragma unroll
    for (int j = 0; j < 4; ++j) s += (float)hrow[lane + 32 * j] * W2[lane + 32 * j];
    #pragma unroll
    for (int m = 1; m < 32; m <<= 1) s += __shfl_xor(s, m, 32);
    if (lane == 0) {
        int t = row % TT, b = row / TT;
        float mk = (t < (cnt[b] >> 1)) ? 1.0f : 0.0f;
        out[row] = mk / (1.0f + __expf(-(s + b2[0])));
    }
}

__global__ __launch_bounds__(256)
void vote_reduce(const float* __restrict__ vals, const int* __restrict__ cnt,
                 float* __restrict__ out)
{
    __shared__ float sh[256];
    int b = blockIdx.x;
    float s = 0.0f;
    for (int t = threadIdx.x; t < TT; t += 256) s += vals[b * TT + t];
    sh[threadIdx.x] = s;
    __syncthreads();
    for (int st = 128; st > 0; st >>= 1) {
        if (threadIdx.x < st) sh[threadIdx.x] += sh[threadIdx.x + st];
        __syncthreads();
    }
    if (threadIdx.x == 0) {
        int c = cnt[b] >> 1; if (c > TT) c = TT;
        out[b] = sh[0] / fmaxf((float)c, 1.0f);
    }
}

// ---------------------------------------------------------------------------
// host side
// ---------------------------------------------------------------------------
static inline void gemm_launch(hipStream_t stream,
                               const __bf16* A, long long aB, int lda, int tA,
                               const __bf16* W, long long wB, int ldw,
                               const float* bias,
                               __bf16* C, long long cB, int ldc,
                               int M, int N, int K, int act, int batch)
{
    dim3 grid(N / 128, M / 128, batch);
    gemm_wmma<<<grid, dim3(256), 0, stream>>>(A, aB, lda, tA, W, wB, ldw, bias,
                                              C, cB, ldc, M, N, K, act);
}

extern "C" void kernel_launch(void* const* d_in, const int* in_sizes, int n_in,
                              void* d_out, int out_size, void* d_ws, size_t ws_size,
                              hipStream_t stream)
{
    (void)in_sizes; (void)n_in; (void)out_size; (void)ws_size;
    const float* Adj   = (const float*)d_in[0];     // [B,L,NC]
    const int*   cnt   = (const int*)d_in[1];       // [B]
    const float* Linit = (const float*)d_in[2];
    const float* Cinit = (const float*)d_in[3];
    const float* LC_W1 = (const float*)d_in[4];  const float* LC_b1 = (const float*)d_in[5];
    const float* LC_W2 = (const float*)d_in[6];  const float* LC_b2 = (const float*)d_in[7];
    const float* LC_W3 = (const float*)d_in[8];  const float* LC_b3 = (const float*)d_in[9];
    const float* CL_W1 = (const float*)d_in[10]; const float* CL_b1 = (const float*)d_in[11];
    const float* CL_W2 = (const float*)d_in[12]; const float* CL_b2 = (const float*)d_in[13];
    const float* CL_W3 = (const float*)d_in[14]; const float* CL_b3 = (const float*)d_in[15];
    const float* Lu_Wih = (const float*)d_in[16];
    const float* Lu_Whh = (const float*)d_in[17];
    const float* Lu_lni_g = (const float*)d_in[18]; const float* Lu_lni_b = (const float*)d_in[19];
    const float* Lu_lnh_g = (const float*)d_in[20]; const float* Lu_lnh_b = (const float*)d_in[21];
    const float* Lu_lnc_g = (const float*)d_in[22]; const float* Lu_lnc_b = (const float*)d_in[23];
    const float* Cu_Wih = (const float*)d_in[24];
    const float* Cu_Whh = (const float*)d_in[25];
    const float* Cu_lni_g = (const float*)d_in[26]; const float* Cu_lni_b = (const float*)d_in[27];
    const float* Cu_lnh_g = (const float*)d_in[28]; const float* Cu_lnh_b = (const float*)d_in[29];
    const float* Cu_lnc_g = (const float*)d_in[30]; const float* Cu_lnc_b = (const float*)d_in[31];
    const float* att_Wk = (const float*)d_in[32]; const float* att_bk = (const float*)d_in[33];
    const float* att_Wq = (const float*)d_in[34]; const float* att_bq = (const float*)d_in[35];
    const float* att_Wv = (const float*)d_in[36]; const float* att_bv = (const float*)d_in[37];
    const float* asn_W1 = (const float*)d_in[38]; const float* asn_b1 = (const float*)d_in[39];
    const float* asn_W2 = (const float*)d_in[40]; const float* asn_b2 = (const float*)d_in[41];
    const float* vote_W1 = (const float*)d_in[42]; const float* vote_b1 = (const float*)d_in[43];
    const float* vote_W2 = (const float*)d_in[44]; const float* vote_b2 = (const float*)d_in[45];
    float* out = (float*)d_out;

    // -------- workspace layout (bytes, 256B aligned chunks) --------
    char* base = (char*)d_ws;
    size_t off = 0;
    auto alloc = [&](size_t bytes) {
        void* p = base + off;
        off += (bytes + 255) & ~(size_t)255;
        return p;
    };
    __bf16* AdjH = (__bf16*)alloc((size_t)BB * LLIT * NCL * 2);   // 64 MB, L2-resident
    __bf16* Lh   = (__bf16*)alloc((size_t)BB * LLIT * DD * 2);
    float*  Lc   = (float*) alloc((size_t)BB * LLIT * DD * 4);
    __bf16* Ch   = (__bf16*)alloc((size_t)BB * NCL * DD * 2);
    float*  Cc   = (float*) alloc((size_t)BB * NCL * DD * 4);
    __bf16* T1   = (__bf16*)alloc((size_t)BB * NCL * DD * 2);
    __bf16* T2   = (__bf16*)alloc((size_t)BB * NCL * DD * 2);
    __bf16* MSG  = (__bf16*)alloc((size_t)BB * NCL * DD * 2);
    __bf16* XL   = (__bf16*)alloc((size_t)BB * LLIT * CC2 * 2);
    __bf16* Gb1  = (__bf16*)alloc((size_t)BB * NCL * G4 * 2);
    __bf16* Gb2  = (__bf16*)alloc((size_t)BB * NCL * G4 * 2);
    __bf16* RD   = (__bf16*)alloc((size_t)BB * TT * CC2 * 2);
    __bf16* Qb   = (__bf16*)alloc((size_t)BB * TT * CC2 * 2);
    __bf16* Kb   = (__bf16*)alloc((size_t)BB * TT * CC2 * 2);
    __bf16* Vb   = (__bf16*)alloc((size_t)BB * TT * CC2 * 2);
    __bf16* Yb   = (__bf16*)alloc((size_t)BB * TT * CC2 * 2);
    __bf16* H1   = (__bf16*)alloc((size_t)BB * TT * DD * 2);
    float*  VV   = (float*) alloc((size_t)BB * TT * 4);

    // weight conversion helper (f32 -> bf16, once per launch)
    auto cvtw = [&](const float* s, size_t n) {
        __bf16* d = (__bf16*)alloc(n * 2);
        cvt_f32_bf16<<<dim3((unsigned)((n + 255) / 256)), dim3(256), 0, stream>>>(s, d, (long long)n);
        return d;
    };
    __bf16* wLC1 = cvtw(LC_W1, (size_t)DD * DD);
    __bf16* wLC2 = cvtw(LC_W2, (size_t)DD * DD);
    __bf16* wLC3 = cvtw(LC_W3, (size_t)DD * DD);
    __bf16* wCL1 = cvtw(CL_W1, (size_t)DD * DD);
    __bf16* wCL2 = cvtw(CL_W2, (size_t)DD * DD);
    __bf16* wCL3 = cvtw(CL_W3, (size_t)DD * DD);
    __bf16* wLih = cvtw(Lu_Wih, (size_t)CC2 * G4);
    __bf16* wLhh = cvtw(Lu_Whh, (size_t)DD * G4);
    __bf16* wCih = cvtw(Cu_Wih, (size_t)DD * G4);
    __bf16* wChh = cvtw(Cu_Whh, (size_t)DD * G4);
    __bf16* wWk  = cvtw(att_Wk, (size_t)CC2 * CC2);
    __bf16* wWq  = cvtw(att_Wq, (size_t)CC2 * CC2);
    __bf16* wWv  = cvtw(att_Wv, (size_t)CC2 * CC2);
    __bf16* wA1  = cvtw(asn_W1, (size_t)CC2 * DD);
    __bf16* wV1  = cvtw(vote_W1, (size_t)CC2 * DD);

    const int rowsL = BB * LLIT;   // 8192
    const int rowsC = BB * NCL;    // 16384
    const int rowsT = BB * TT;     // 4096

    // -------- convert adjacency once (exact in bf16), init state --------
    {
        long long n = (long long)BB * LLIT * NCL;
        cvt_f32_bf16<<<dim3((unsigned)(n / 256)), dim3(256), 0, stream>>>(Adj, AdjH, n);
    }
    init_state<<<dim3((BB * NCL * DD) / 256), dim3(256), 0, stream>>>(Linit, Cinit, Lh, Lc, Ch, Cc);

    for (int it = 0; it < ITERS; ++it) {
        // literal mlp3 -> T1
        gemm_launch(stream, Lh, 0, DD, 0, wLC1, 0, DD, LC_b1, T1, 0, DD, rowsL, DD, DD, 1, 1);
        gemm_launch(stream, T1, 0, DD, 0, wLC2, 0, DD, LC_b2, T2, 0, DD, rowsL, DD, DD, 1, 1);
        gemm_launch(stream, T2, 0, DD, 0, wLC3, 0, DD, LC_b3, T1, 0, DD, rowsL, DD, DD, 0, 1);
        // LC = A^T @ T1   (batched, transA)
        gemm_launch(stream, AdjH, (long long)LLIT * NCL, NCL, 1,
                    T1, (long long)LLIT * DD, DD, nullptr,
                    MSG, (long long)NCL * DD, DD, NCL, DD, LLIT, 0, BB);
        // clause LSTM gates
        gemm_launch(stream, MSG, 0, DD, 0, wCih, 0, G4, nullptr, Gb1, 0, G4, rowsC, G4, DD, 0, 1);
        gemm_launch(stream, Ch,  0, DD, 0, wChh, 0, G4, nullptr, Gb2, 0, G4, rowsC, G4, DD, 0, 1);
        lstm_ln<<<dim3(rowsC / 8), dim3(256), 0, stream>>>(Gb1, Gb2,
            Cu_lni_g, Cu_lni_b, Cu_lnh_g, Cu_lnh_b, Cu_lnc_g, Cu_lnc_b, Cc, Ch, Cc);
        // clause mlp3 -> T1
        gemm_launch(stream, Ch, 0, DD, 0, wCL1, 0, DD, CL_b1, T1, 0, DD, rowsC, DD, DD, 1, 1);
        gemm_launch(stream, T1, 0, DD, 0, wCL2, 0, DD, CL_b2, T2, 0, DD, rowsC, DD, DD, 1, 1);
        gemm_launch(stream, T2, 0, DD, 0, wCL3, 0, DD, CL_b3, T1, 0, DD, rowsC, DD, DD, 0, 1);
        // CL = A @ T1   (batched)
        gemm_launch(stream, AdjH, (long long)LLIT * NCL, NCL, 0,
                    T1, (long long)NCL * DD, DD, nullptr,
                    MSG, (long long)LLIT * DD, DD, LLIT, DD, NCL, 0, BB);
        // literal LSTM
        build_xl<<<dim3((BB * LLIT * DD) / 256), dim3(256), 0, stream>>>(MSG, Lh, XL);
        gemm_launch(stream, XL, 0, CC2, 0, wLih, 0, G4, nullptr, Gb1, 0, G4, rowsL, G4, CC2, 0, 1);
        gemm_launch(stream, Lh, 0, DD,  0, wLhh, 0, G4, nullptr, Gb2, 0, G4, rowsL, G4, DD,  0, 1);
        lstm_ln<<<dim3(rowsL / 8), dim3(256), 0, stream>>>(Gb1, Gb2,
            Lu_lni_g, Lu_lni_b, Lu_lnh_g, Lu_lnh_b, Lu_lnc_g, Lu_lnc_b, Lc, Lh, Lc);
    }

    // -------- readout + attention --------
    build_read<<<dim3((BB * TT * DD) / 256), dim3(256), 0, stream>>>(Lh, cnt, RD);
    gemm_launch(stream, RD, 0, CC2, 0, wWk, 0, CC2, att_bk, Kb, 0, CC2, rowsT, CC2, CC2, 0, 1);
    gemm_launch(stream, RD, 0, CC2, 0, wWq, 0, CC2, att_bq, Qb, 0, CC2, rowsT, CC2, CC2, 0, 1);
    gemm_launch(stream, RD, 0, CC2, 0, wWv, 0, CC2, att_bv, Vb, 0, CC2, rowsT, CC2, CC2, 0, 1);
    attn_flash<<<dim3(TT / (16 * ATTN_WAVES), 8, BB), dim3(32 * ATTN_WAVES), 0, stream>>>(Qb, Kb, Vb, cnt, Yb);

    // -------- heads --------
    gemm_launch(stream, Yb, 0, CC2, 0, wA1, 0, DD, asn_b1, H1, 0, DD, rowsT, DD, CC2, 1, 1);
    head_out<<<dim3(rowsT / 8), dim3(256), 0, stream>>>(H1, asn_W2, asn_b2, cnt, out + BB);
    gemm_launch(stream, Yb, 0, CC2, 0, wV1, 0, DD, vote_b1, H1, 0, DD, rowsT, DD, CC2, 1, 1);
    head_out<<<dim3(rowsT / 8), dim3(256), 0, stream>>>(H1, vote_W2, vote_b2, cnt, VV);
    vote_reduce<<<dim3(BB), dim3(256), 0, stream>>>(VV, cnt, out);
}
